// N_aryTreeLSTM_88802743812223
// MI455X (gfx1250) — compile-verified, matrix operand order checked by reading, not run on hardware
//
#include <hip/hip_runtime.h>

#define IN_DIMX 300
#define KPAD    320
#define MEMD    256
#define OUTD    1024
#define DEPTHX  16
#define LEAVESX 65536

typedef __attribute__((ext_vector_type(16))) __bf16 v16bf;
typedef __attribute__((ext_vector_type(8)))  __bf16 v8bf;
typedef __attribute__((ext_vector_type(8)))  float  v8f;

// K index inside a 16x32 bf16 WMMA A/B fragment for element e (0..15), half h (lane>>4)
__device__ __forceinline__ int wmma_k(int e, int h) {
  return (e & 7) + (h << 3) + ((e >> 3) << 4);
}
// inverse map: within-chunk k (0..31) -> (element, half)
__device__ __forceinline__ void wmma_k_inv(int kk, int& e, int& h) {
  h = (kk >> 3) & 1;
  e = (kk & 7) | ((kk >> 4) << 3);
}
__device__ __forceinline__ float sigf(float x) { return 1.0f / (1.0f + __expf(-x)); }

#define ALANE 24  // bf16 elements per lane per A-fragment (48B stride -> conflict-free ds_b128)

// LDS A-fragment index for node-row m (0..31), absolute k; KC = k-chunks per M-tile
__device__ __forceinline__ int a_idx(int m, int k, int KC) {
  int e, h;
  wmma_k_inv(k & 31, e, h);
  int lane = (h << 4) | (m & 15);
  return (((m >> 4) * KC + (k >> 5)) * 32 + lane) * ALANE + e;
}

__device__ __forceinline__ v16bf load_a(const __bf16* p) {
  v8bf lo = *(const v8bf*)p;
  v8bf hi = *(const v8bf*)(p + 8);
  return __builtin_shufflevector(lo, hi, 0, 1, 2, 3, 4, 5, 6, 7,
                                 8, 9, 10, 11, 12, 13, 14, 15);
}

// ---------------- prep: weights -> pre-swizzled bf16 WMMA B-fragments ----------------
// W_fx [256 x 300] -> frags [16 tiles][10 kc][32 lanes][16 e], K zero-padded to 320
__global__ void prep_wfx(const float* __restrict__ W, __bf16* __restrict__ F) {
  int idx = blockIdx.x * blockDim.x + threadIdx.x;
  if (idx >= 16 * 10 * 512) return;
  int e    = idx & 15;
  int lane = (idx >> 4) & 31;
  int kc   = (idx >> 9) % 10;
  int tile = (idx >> 9) / 10;
  int n = tile * 16 + (lane & 15);
  int k = kc * 32 + wmma_k(e, lane >> 4);
  float v = (k < IN_DIMX) ? W[n * IN_DIMX + k] : 0.0f;
  F[idx] = (__bf16)v;
}

// combined [W_iouh;W_fh] as B [256 x 1024] -> frags [64 tiles][8 kc][32 lanes][16 e]
__global__ void prep_wc(const float* __restrict__ Wiouh, const float* __restrict__ Wfh,
                        __bf16* __restrict__ F) {
  int idx = blockIdx.x * blockDim.x + threadIdx.x;
  if (idx >= 64 * 8 * 512) return;
  int e    = idx & 15;
  int lane = (idx >> 4) & 31;
  int kc   = (idx >> 9) & 7;
  int tile = idx >> 12;
  int col = tile * 16 + (lane & 15);
  int k   = kc * 32 + wmma_k(e, lane >> 4);
  float v = (col < 768) ? Wiouh[col * MEMD + k] : Wfh[(col - 768) * MEMD + k];
  F[idx] = (__bf16)v;
}

__global__ void prep_bias(const float* __restrict__ bfx_in, const float* __restrict__ biouh,
                          const float* __restrict__ bfh, float* __restrict__ bfx_out,
                          float* __restrict__ bc_out) {
  int i = blockIdx.x * blockDim.x + threadIdx.x;
  if (i < MEMD) bfx_out[i] = bfx_in[i];
  if (i < OUTD) bc_out[i] = (i < 768) ? 2.0f * biouh[i] : 2.0f * bfh[i - 768];
}

// ---------------- leaf pass: 32 leaves/block; fx = x @ W_fx^T + b ----------------
__global__ __launch_bounds__(256) void leaf_kernel(
    const float* __restrict__ x, const __bf16* __restrict__ WF,
    const float* __restrict__ bfx, float* __restrict__ c_out, float* __restrict__ h_out) {
  __shared__ __bf16 s_a[2 * 10 * 32 * ALANE];  // 30KB: A-frags for both M-halves
  const int tid = threadIdx.x, lane = tid & 31, wave = tid >> 5;
  const int node0 = blockIdx.x * 32;

  // stage inputs straight into swizzled bf16 A-fragments (K padded 300 -> 320)
  for (int m = wave; m < 32; m += 8) {
    const float* xr = x + (size_t)(node0 + m) * IN_DIMX;
    for (int k = lane; k < KPAD; k += 32)
      s_a[a_idx(m, k, 10)] = (__bf16)((k < IN_DIMX) ? xr[k] : 0.0f);
  }
  __syncthreads();

  const v8f vzero = {};
  v8f acc[2][2];  // [tt][mh]
#pragma unroll
  for (int tt = 0; tt < 2; ++tt)
#pragma unroll
    for (int mh = 0; mh < 2; ++mh) acc[tt][mh] = vzero;

#pragma unroll
  for (int kc = 0; kc < 10; ++kc) {
    const v16bf a0 = load_a(s_a + ((0 * 10 + kc) * 32 + lane) * ALANE);
    const v16bf a1 = load_a(s_a + ((1 * 10 + kc) * 32 + lane) * ALANE);
#pragma unroll
    for (int tt = 0; tt < 2; ++tt) {
      const int tile = wave * 2 + tt;
      v16bf b = *(const v16bf*)(WF + (((size_t)tile * 10 + kc) * 32 + lane) * 16);
      acc[tt][0] = __builtin_amdgcn_wmma_f32_16x16x32_bf16(false, a0, false, b,
                                                           (short)0, acc[tt][0], false, false);
      acc[tt][1] = __builtin_amdgcn_wmma_f32_16x16x32_bf16(false, a1, false, b,
                                                           (short)0, acc[tt][1], false, false);
    }
  }

  const int n = lane & 15, mb = (lane >> 4) * 8;
#pragma unroll
  for (int tt = 0; tt < 2; ++tt) {
    const int chan = (wave * 2 + tt) * 16 + n;
    const float bv = bfx[chan];
#pragma unroll
    for (int mh = 0; mh < 2; ++mh) {
#pragma unroll
      for (int r = 0; r < 8; ++r) {
        const int node = node0 + mh * 16 + mb + r;
        const float fx = acc[tt][mh][r] + bv;
        c_out[(size_t)node * MEMD + chan] = fx;
        h_out[(size_t)node * MEMD + chan] = sigf(fx) * tanhf(fx);
      }
    }
  }
}

// ---------------- one tree level: 32 parents/block from child pairs (2i, 2i+1) ----------------
#define HSS 260  // padded LDS row stride (floats) for csum
__global__ __launch_bounds__(256) void tree_kernel(
    const float* __restrict__ c_in, const float* __restrict__ h_in,
    float* __restrict__ c_out, float* __restrict__ h_out,
    const __bf16* __restrict__ WF, const float* __restrict__ bc, int m_out) {
  __shared__ __bf16 s_a[2 * 8 * 32 * ALANE];  // 24KB: hsum A-frags, both M-halves
  __shared__ float s_cs[32 * HSS];            // 33KB: csum for epilogue
  const int tid = threadIdx.x, lane = tid & 31, wave = tid >> 5;
  const int node0 = blockIdx.x * 32;

  for (int m = wave; m < 32; m += 8) {
    const int node = node0 + m;
    if (node < m_out) {
      const float* hp = h_in + (size_t)(2 * node) * MEMD;  // children rows adjacent
      const float* cp = c_in + (size_t)(2 * node) * MEMD;
      for (int k = lane; k < MEMD; k += 32) {
        s_a[a_idx(m, k, 8)] = (__bf16)(hp[k] + hp[MEMD + k]);
        s_cs[m * HSS + k] = cp[k] + cp[MEMD + k];
      }
    } else {
      for (int k = lane; k < MEMD; k += 32) {
        s_a[a_idx(m, k, 8)] = (__bf16)0.0f;
        s_cs[m * HSS + k] = 0.0f;
      }
    }
  }
  __syncthreads();

  // wave w owns channels [32w,32w+32) of ALL four gates -> epilogue stays in registers
  const v8f vzero = {};
  v8f acc[4][2][2];  // [gate][tt][mh]
#pragma unroll
  for (int g = 0; g < 4; ++g)
#pragma unroll
    for (int tt = 0; tt < 2; ++tt)
#pragma unroll
      for (int mh = 0; mh < 2; ++mh) acc[g][tt][mh] = vzero;

#pragma unroll
  for (int kc = 0; kc < 8; ++kc) {
    const v16bf a0 = load_a(s_a + ((0 * 8 + kc) * 32 + lane) * ALANE);
    const v16bf a1 = load_a(s_a + ((1 * 8 + kc) * 32 + lane) * ALANE);
#pragma unroll
    for (int g = 0; g < 4; ++g) {
#pragma unroll
      for (int tt = 0; tt < 2; ++tt) {
        const int tile = g * 16 + wave * 2 + tt;
        v16bf b = *(const v16bf*)(WF + (((size_t)tile * 8 + kc) * 32 + lane) * 16);
        acc[g][tt][0] = __builtin_amdgcn_wmma_f32_16x16x32_bf16(false, a0, false, b,
                                                                (short)0, acc[g][tt][0], false, false);
        acc[g][tt][1] = __builtin_amdgcn_wmma_f32_16x16x32_bf16(false, a1, false, b,
                                                                (short)0, acc[g][tt][1], false, false);
      }
    }
  }

  const int n = lane & 15, mb = (lane >> 4) * 8;
#pragma unroll
  for (int tt = 0; tt < 2; ++tt) {
    const int chan = wave * 32 + tt * 16 + n;
    const float bi = bc[chan], bo = bc[MEMD + chan];
    const float bu = bc[2 * MEMD + chan], bf = bc[3 * MEMD + chan];
#pragma unroll
    for (int mh = 0; mh < 2; ++mh) {
#pragma unroll
      for (int r = 0; r < 8; ++r) {
        const int m = mh * 16 + mb + r, node = node0 + m;
        if (node < m_out) {
          const float cs = s_cs[m * HSS + chan];
          const float cn = sigf(acc[0][tt][mh][r] + bi) * tanhf(acc[2][tt][mh][r] + bu)
                         + sigf(acc[3][tt][mh][r] + bf) * cs;
          c_out[(size_t)node * MEMD + chan] = cn;
          h_out[(size_t)node * MEMD + chan] = sigf(acc[1][tt][mh][r] + bo) * tanhf(cn);
        }
      }
    }
  }
}

__global__ void copy_root(const float* __restrict__ c, const float* __restrict__ h,
                          float* __restrict__ out) {
  const int t = threadIdx.x;
  out[t] = c[t];
  out[MEMD + t] = h[t];
}

extern "C" void kernel_launch(void* const* d_in, const int* in_sizes, int n_in,
                              void* d_out, int out_size, void* d_ws, size_t ws_size,
                              hipStream_t stream) {
  const float* x      = (const float*)d_in[0];
  const float* W_fx   = (const float*)d_in[1];
  const float* b_fx   = (const float*)d_in[2];
  const float* W_iouh = (const float*)d_in[3];
  const float* b_iouh = (const float*)d_in[4];
  const float* W_fh   = (const float*)d_in[5];
  const float* b_fh   = (const float*)d_in[6];

  char* ws = (char*)d_ws;
  size_t off = 0;
  float* cA = (float*)(ws + off); off += (size_t)LEAVESX * MEMD * 4;
  float* hA = (float*)(ws + off); off += (size_t)LEAVESX * MEMD * 4;
  float* cB = (float*)(ws + off); off += (size_t)(LEAVESX / 2) * MEMD * 4;
  float* hB = (float*)(ws + off); off += (size_t)(LEAVESX / 2) * MEMD * 4;
  __bf16* WfxF = (__bf16*)(ws + off); off += (size_t)16 * 10 * 512 * 2;
  __bf16* WcF  = (__bf16*)(ws + off); off += (size_t)64 * 8 * 512 * 2;
  float* bfxW = (float*)(ws + off); off += MEMD * 4;
  float* bcW  = (float*)(ws + off); off += OUTD * 4;

  prep_wfx<<<(16 * 10 * 512 + 255) / 256, 256, 0, stream>>>(W_fx, WfxF);
  prep_wc<<<(64 * 8 * 512 + 255) / 256, 256, 0, stream>>>(W_iouh, W_fh, WcF);
  prep_bias<<<(OUTD + 255) / 256, 256, 0, stream>>>(b_fx, b_iouh, b_fh, bfxW, bcW);

  leaf_kernel<<<LEAVESX / 32, 256, 0, stream>>>(x, WfxF, bfxW, cA, hA);

  const float* ci = cA;
  const float* hi = hA;
  int m = LEAVESX;
  for (int lvl = 0; lvl < DEPTHX; ++lvl) {
    m >>= 1;
    float* co = (lvl & 1) ? cA : cB;
    float* ho = (lvl & 1) ? hA : hB;
    tree_kernel<<<(m + 31) / 32, 256, 0, stream>>>(ci, hi, co, ho, WcF, bcW, m);
    ci = co; hi = ho;
  }

  copy_root<<<1, MEMD, 0, stream>>>(ci, hi, (float*)d_out);

  (void)in_sizes; (void)n_in; (void)out_size; (void)ws_size;
}
